// LinearMultiHeadAttn_15891378995457
// MI455X (gfx1250) — matrix operand order repaired and checked by reading, not compiled, not run on hardware
//
#include <hip/hip_runtime.h>
#include <hip/hip_bf16.h>

// ---------------- problem constants ----------------
#define S_LEN   2048
#define BATCH   2
#define DMODEL  512
#define NHEAD   8
#define DHEAD   64
#define ROWS    (S_LEN * BATCH)        // 4096
#define CHUNK   64
#define NCHUNK  (S_LEN / CHUNK)        // 32
#define BH      (BATCH * NHEAD)        // 16
#define STATE_SZ (DHEAD * DHEAD + DHEAD)   // 4160 floats per chunk state
#define QK_SCALE 0.125f                // 1/sqrt(64)
#define EPS_DEN  1e-5f
#define LNEPS    1e-5f

typedef __attribute__((ext_vector_type(16))) __bf16 v16bf;
typedef __attribute__((ext_vector_type(8)))  __bf16 v8bf;
typedef __attribute__((ext_vector_type(8)))  float  v8f;

static __device__ inline v8f wmma_bf16(v16bf a, v16bf b, v8f c) {
  return __builtin_amdgcn_wmma_f32_16x16x32_bf16(
      /*neg_a=*/false, a, /*neg_b=*/false, b,
      /*c_mod=*/(short)0, c, /*reuse_a=*/false, /*reuse_b=*/false);
}

static __device__ inline float elu1(float x) {
  return x > 0.f ? x + 1.f : __expf(x);
}

// ---- CDNA5 async global->LDS copy (16B per lane), tracked by ASYNCcnt ----
static __device__ inline void async_copy_b128(void* lds_dst, const void* gsrc) {
  unsigned ldsoff = (unsigned)(uintptr_t)lds_dst;   // low 32 bits = LDS offset
  asm volatile("global_load_async_to_lds_b128 %0, %1, off"
               :: "v"(ldsoff), "v"(gsrc) : "memory");
}
static __device__ inline void async_wait0() {
  asm volatile("s_wait_asynccnt 0" ::: "memory");
}

// ---- fragment loaders (layouts per CDNA5 ISA 7.12.2) ----
// A 16x32 (MxK): lane<16 -> M=lane, K = k0+{0..7,16..23}; lane>=16 -> M=lane-16, K = k0+{8..15,24..31}
static __device__ inline v16bf a_from_f32(const float* base, int ld, int m0, int k0, int lane) {
  const int row = m0 + (lane & 15);
  const int kh  = (lane >> 4) * 8;
  const float* p = base + (size_t)row * ld + k0 + kh;
  v16bf a;
#pragma unroll
  for (int i = 0; i < 8; ++i) { a[i] = (__bf16)p[i]; a[i + 8] = (__bf16)p[i + 16]; }
  return a;
}
static __device__ inline v16bf a_from_bf16(const __bf16* base, int ld, int m0, int k0, int lane) {
  const int row = m0 + (lane & 15);
  const int kh  = (lane >> 4) * 8;
  const __bf16* p = base + (size_t)row * ld + k0 + kh;
  const v8bf lo = *(const v8bf*)p;
  const v8bf hi = *(const v8bf*)(p + 16);
  v16bf a;
#pragma unroll
  for (int i = 0; i < 8; ++i) { a[i] = lo[i]; a[i + 8] = hi[i]; }
  return a;
}
// A transposed source: A[m][k] = src[k][m]  (for K^T V) -- strided, per element
static __device__ inline v16bf aT_from_bf16(const __bf16* base, int ld, int m0, int k0, int lane) {
  const int m  = m0 + (lane & 15);
  const int kh = (lane >> 4) * 8;
  v16bf a;
#pragma unroll
  for (int i = 0; i < 8; ++i) {
    a[i]     = base[(size_t)(k0 + kh + i) * ld + m];
    a[i + 8] = base[(size_t)(k0 + 16 + kh + i) * ld + m];
  }
  return a;
}
// B 32x16 (KxN) from f32 column-strided source
static __device__ inline v16bf b_from_f32(const float* base, int ld, int k0, int n0, int lane) {
  const int col = n0 + (lane & 15);
  const int ks  = k0 + (lane >> 4) * 8;
  v16bf b;
#pragma unroll
  for (int i = 0; i < 16; ++i) b[i] = (__bf16)base[(size_t)(ks + i + (i >= 8 ? 8 : 0)) * ld + col];
  return b;
}
// B 32x16 where source is stored N-major: B[k][n] = src[n][k] -> two 16B contiguous runs per lane
static __device__ inline v16bf bT_from_bf16(const __bf16* base, int ld, int k0, int n0, int lane) {
  const int col = n0 + (lane & 15);
  const int ks  = k0 + (lane >> 4) * 8;
  const __bf16* p = base + (size_t)col * ld + ks;
  const v8bf lo = *(const v8bf*)p;
  const v8bf hi = *(const v8bf*)(p + 16);
  v16bf b;
#pragma unroll
  for (int i = 0; i < 8; ++i) { b[i] = lo[i]; b[i + 8] = hi[i]; }
  return b;
}

// =====================================================================
// Kernel 1: fused QKV projection. out cols [0,512)=Q (elu+1)*SCALE,
// [512,1024)=K elu+1, [1024,1536)=V raw.
// Q,K stored bf16 [b,h,s,d]; V stored TRANSPOSED bf16 [b,h,d,s] so that
// later V B-fragments are contiguous in LDS.
// =====================================================================
__global__ __launch_bounds__(128) void k_qkv(const float* __restrict__ h,
                                             const float* __restrict__ Wq,
                                             const float* __restrict__ Wkv,
                                             __bf16* __restrict__ Qb,
                                             __bf16* __restrict__ Kb,
                                             __bf16* __restrict__ Vtb) {
  const int lane = threadIdx.x & 31;
  const int wave = threadIdx.x >> 5;
  const int mt   = blockIdx.x % (ROWS / 16);   // 256 row tiles
  const int nt64 = blockIdx.x / (ROWS / 16);   // 24 groups of 64 cols
  const int m0   = mt * 16;
  const int n0   = nt64 * 64 + wave * 16;      // global out col [0,1536) (uniform per wave)

  const float* Wsrc; int ldw, c0;
  if (n0 < DMODEL) { Wsrc = Wq;  ldw = DMODEL;     c0 = n0; }
  else             { Wsrc = Wkv; ldw = 2 * DMODEL; c0 = n0 - DMODEL; }

  v8f acc = {};
  for (int k0 = 0; k0 < DMODEL; k0 += 32) {
    const int kp = (k0 + 32) & (DMODEL - 1);   // branchless wrap for prefetch
    __builtin_prefetch(Wsrc + (size_t)kp * ldw + c0, 0, 1);
    __builtin_prefetch(h + (size_t)m0 * DMODEL + kp, 0, 1);
    v16bf a = a_from_f32(h, DMODEL, m0, k0, lane);
    v16bf b = b_from_f32(Wsrc, ldw, k0, c0, lane);
    acc = wmma_bf16(a, b, acc);
  }

  // wave-uniform epilogue routing (tile never straddles a Q/K/V boundary)
  const int part = n0 >> 9;            // 0=q 1=k 2=v  (scalar)
  const int cp0  = n0 & 511;           // scalar
  const int hI   = cp0 >> 6;           // scalar
  const int d0   = cp0 & 63;           // scalar tile base within head

  const int n  = lane & 15;
  const int hi = lane >> 4;
  const int d  = d0 + n;
#pragma unroll
  for (int r = 0; r < 8; ++r) {
    const int m  = m0 + r + 8 * hi;          // row = s*BATCH + b
    const int b_ = m & (BATCH - 1);
    const int s  = m >> 1;
    const int bh = b_ * NHEAD + hI;
    float v = acc[r];
    if (part == 0) {
      Qb[((size_t)bh * S_LEN + s) * DHEAD + d] = (__bf16)(elu1(v) * QK_SCALE);
    } else if (part == 1) {
      Kb[((size_t)bh * S_LEN + s) * DHEAD + d] = (__bf16)elu1(v);
    } else {
      Vtb[((size_t)bh * DHEAD + d) * S_LEN + s] = (__bf16)v;   // transposed
    }
  }
}

// =====================================================================
// Kernel 2: per-chunk states  S_c = K_c^T V_c  (64x64, stored row-major
// [dk][dv]) and ksum_c (64). One workgroup (4 waves) per (bh, chunk).
// =====================================================================
__global__ __launch_bounds__(128) void k_chunk_state(const __bf16* __restrict__ Kb,
                                                     const __bf16* __restrict__ Vtb,
                                                     float* __restrict__ stats) {
  __shared__ __bf16 Kl[CHUNK * DHEAD];     // [j][dk]
  __shared__ __bf16 Vt[DHEAD * CHUNK];     // [dv][j]  (transposed tile)
  const int tid = threadIdx.x, lane = tid & 31, wave = tid >> 5;
  const int bh = blockIdx.x / NCHUNK;
  const int c  = blockIdx.x % NCHUNK;

  const size_t kbase = ((size_t)bh * S_LEN + (size_t)c * CHUNK) * DHEAD;
  for (int i = tid; i < CHUNK * DHEAD / 8; i += 128) {
    async_copy_b128(Kl + i * 8, Kb + kbase + (size_t)i * 8);
    // V^T tile: row dv = i>>3, 8-elem segment (i&7) at column c*64
    const int dv = i >> 3, seg = i & 7;
    async_copy_b128(Vt + dv * CHUNK + seg * 8,
                    Vtb + ((size_t)bh * DHEAD + dv) * S_LEN + c * CHUNK + seg * 8);
  }
  async_wait0();
  __syncthreads();

  float* st = stats + (size_t)blockIdx.x * STATE_SZ;
  const int i0 = wave * 16;                 // d_k strip
  const int n = lane & 15, hi = lane >> 4;
#pragma unroll
  for (int nt = 0; nt < 4; ++nt) {
    v8f acc = {};
#pragma unroll
    for (int kk = 0; kk < 2; ++kk) {
      v16bf a = aT_from_bf16(Kl, DHEAD, i0, kk * 32, lane);     // A[dk][j] = K[j][dk]
      v16bf b = bT_from_bf16(Vt, CHUNK, kk * 32, nt * 16, lane); // B[j][dv] = Vt[dv][j]
      acc = wmma_bf16(a, b, acc);
    }
#pragma unroll
    for (int r = 0; r < 8; ++r)
      st[(size_t)(i0 + r + 8 * hi) * DHEAD + nt * 16 + n] = acc[r];
  }
  if (tid < DHEAD) {
    float s = 0.f;
    for (int j = 0; j < CHUNK; ++j) s += (float)Kl[j * DHEAD + tid];
    st[DHEAD * DHEAD + tid] = s;
  }
}

// =====================================================================
// Kernel 3: per-chunk attention output.
//   prefix = sum of states of chunks < c   (exclusive prefix, stored
//   transposed in LDS as SkvT[dv][dk] for contiguous B-fragments)
//   S = Q_c K_c^T (causal masked), num = S@V_c + Q_c@prefixKV,
//   den = rowsum(S) + q . prefix_ksum + eps,  out = num/den  (bf16)
// =====================================================================
__global__ __launch_bounds__(128) void k_attn(const __bf16* __restrict__ Qb,
                                              const __bf16* __restrict__ Kb,
                                              const __bf16* __restrict__ Vtb,
                                              const float* __restrict__ stats,
                                              __bf16* __restrict__ attnb) {
  __shared__ __bf16 SkvT[DHEAD * DHEAD];         // prefix KV state, [dv][dk]
  __shared__ float  ksum[DHEAD];                 // prefix k-sum
  __shared__ __bf16 Ql[CHUNK * DHEAD];           // [i][dk]
  __shared__ __bf16 Kl[CHUNK * DHEAD];           // [j][dk]
  __shared__ __bf16 Vt[DHEAD * CHUNK];           // [dv][j]
  __shared__ __bf16 Pl[4][16 * CHUNK];           // masked scores per wave strip
  __shared__ float  denomI[CHUNK];               // inter-chunk denominator per row

  const int tid = threadIdx.x, lane = tid & 31, wave = tid >> 5;
  const int bh = blockIdx.x / NCHUNK;
  const int c  = blockIdx.x % NCHUNK;

  // kick off async staging of Q,K,V^T chunk tiles first (overlap with prefix math)
  const size_t base = ((size_t)bh * S_LEN + (size_t)c * CHUNK) * DHEAD;
  for (int i = tid; i < CHUNK * DHEAD / 8; i += 128) {
    async_copy_b128(Ql + i * 8, Qb + base + (size_t)i * 8);
    async_copy_b128(Kl + i * 8, Kb + base + (size_t)i * 8);
    const int dv = i >> 3, seg = i & 7;
    async_copy_b128(Vt + dv * CHUNK + seg * 8,
                    Vtb + ((size_t)bh * DHEAD + dv) * S_LEN + c * CHUNK + seg * 8);
  }

  // exclusive prefix over chunk states (overlaps the async copies)
  for (int idx = tid; idx < STATE_SZ; idx += 128) {
    float acc = 0.f;
    for (int cc = 0; cc < c; ++cc)
      acc += stats[((size_t)bh * NCHUNK + cc) * STATE_SZ + idx];
    if (idx < DHEAD * DHEAD)
      SkvT[(idx & 63) * DHEAD + (idx >> 6)] = (__bf16)acc;   // transpose on write
    else
      ksum[idx - DHEAD * DHEAD] = acc;
  }

  async_wait0();
  __syncthreads();

  // inter-chunk denominator: q_i . prefix_ksum  (vectorized 16B LDS reads)
  if (tid < CHUNK) {
    float s = 0.f;
    const v8bf* qv = (const v8bf*)(Ql + tid * DHEAD);
#pragma unroll
    for (int d8 = 0; d8 < 8; ++d8) {
      const v8bf q = qv[d8];
#pragma unroll
      for (int j = 0; j < 8; ++j) s += (float)q[j] * ksum[d8 * 8 + j];
    }
    denomI[tid] = s;
  }
  __syncthreads();

  const int i0 = wave * 16;                     // query-row strip
  const int n = lane & 15, hi = lane >> 4;
  float rsum[8];
#pragma unroll
  for (int r = 0; r < 8; ++r) rsum[r] = 0.f;

  // scores, causal mask, P store, intra row sums
#pragma unroll
  for (int nt = 0; nt < 4; ++nt) {
    v8f sc = {};
#pragma unroll
    for (int kk = 0; kk < 2; ++kk) {
      v16bf a = a_from_bf16(Ql, DHEAD, i0, kk * 32, lane);
      v16bf b = bT_from_bf16(Kl, DHEAD, kk * 32, nt * 16, lane);  // B[dk][j] = K[j][dk]
      sc = wmma_bf16(a, b, sc);
    }
#pragma unroll
    for (int r = 0; r < 8; ++r) {
      const int i_loc = i0 + r + 8 * hi;
      const int j_loc = nt * 16 + n;
      const float v = (j_loc <= i_loc) ? sc[r] : 0.f;
      rsum[r] += v;
      Pl[wave][(r + 8 * hi) * CHUNK + j_loc] = (__bf16)v;
    }
  }
  // row-sum reduce across the 16 lanes of each half (halves hold different rows)
#pragma unroll
  for (int r = 0; r < 8; ++r) {
#pragma unroll
    for (int off = 1; off < 16; off <<= 1)
      rsum[r] += __shfl_xor(rsum[r], off, 32);
  }

  const int b_ = bh / NHEAD, hI = bh % NHEAD;
#pragma unroll
  for (int nt = 0; nt < 4; ++nt) {
    v8f acc = {};
#pragma unroll
    for (int kk = 0; kk < 2; ++kk) {
      // intra: P @ V       (B[j][dv] = Vt[dv][j], contiguous)
      acc = wmma_bf16(a_from_bf16(Pl[wave], CHUNK, 0, kk * 32, lane),
                      bT_from_bf16(Vt, CHUNK, kk * 32, nt * 16, lane), acc);
      // inter: Q @ prefixKV (B[dk][dv] = SkvT[dv][dk], contiguous)
      acc = wmma_bf16(a_from_bf16(Ql, DHEAD, i0, kk * 32, lane),
                      bT_from_bf16(SkvT, DHEAD, kk * 32, nt * 16, lane), acc);
    }
#pragma unroll
    for (int r = 0; r < 8; ++r) {
      const int i_loc = i0 + r + 8 * hi;
      const float den = rsum[r] + denomI[i_loc] + EPS_DEN;
      const float ov  = acc[r] / den;
      const int s_glob = c * CHUNK + i_loc;
      const int row = s_glob * BATCH + b_;          // [S,B] flattened
      attnb[(size_t)row * DMODEL + hI * DHEAD + nt * 16 + n] = (__bf16)ov;
    }
  }
}

// =====================================================================
// Kernel 4: out = LayerNorm(h + attn @ Wo).  One block per 16 rows,
// 256 threads (8 waves), each wave owns a 64-col range (4 WMMA tiles).
// =====================================================================
__global__ __launch_bounds__(256) void k_proj_ln(const __bf16* __restrict__ attnb,
                                                 const float* __restrict__ Wo,
                                                 const float* __restrict__ h,
                                                 const float* __restrict__ gamma,
                                                 const float* __restrict__ beta,
                                                 float* __restrict__ out) {
  __shared__ float X[16 * DMODEL];
  const int tid = threadIdx.x, lane = tid & 31, wave = tid >> 5;
  const int m0 = blockIdx.x * 16;
  const int nbase = wave * 64;

  v8f acc[4] = { {}, {}, {}, {} };
  for (int k0 = 0; k0 < DMODEL; k0 += 32) {
    const int kp = (k0 + 32) & (DMODEL - 1);
    __builtin_prefetch(Wo + (size_t)kp * DMODEL + nbase, 0, 1);
    v16bf a = a_from_bf16(attnb, DMODEL, m0, k0, lane);
#pragma unroll
    for (int nt = 0; nt < 4; ++nt) {
      v16bf b = b_from_f32(Wo, DMODEL, k0, nbase + nt * 16, lane);
      acc[nt] = wmma_bf16(a, b, acc[nt]);
    }
  }
  const int n = lane & 15, hi = lane >> 4;
#pragma unroll
  for (int nt = 0; nt < 4; ++nt)
#pragma unroll
    for (int r = 0; r < 8; ++r)
      X[(r + 8 * hi) * DMODEL + nbase + nt * 16 + n] = acc[nt][r];
  __syncthreads();

  // LayerNorm: 16 threads per row, 32 cols each; halves of a wave hold 2 rows
  const int row = tid >> 4;
  const int seg = tid & 15;
  const int gm  = m0 + row;
  float xs[32];
  float sum = 0.f;
#pragma unroll
  for (int i = 0; i < 32; ++i) {
    const int col = seg * 32 + i;
    const float xv = X[row * DMODEL + col] + h[(size_t)gm * DMODEL + col];
    xs[i] = xv;
    sum += xv;
  }
#pragma unroll
  for (int off = 1; off < 16; off <<= 1) sum += __shfl_xor(sum, off, 32);
  const float mu = sum * (1.f / DMODEL);
  float vs = 0.f;
#pragma unroll
  for (int i = 0; i < 32; ++i) { const float d = xs[i] - mu; vs += d * d; }
#pragma unroll
  for (int off = 1; off < 16; off <<= 1) vs += __shfl_xor(vs, off, 32);
  const float rstd = rsqrtf(vs * (1.f / DMODEL) + LNEPS);
#pragma unroll
  for (int i = 0; i < 32; ++i) {
    const int col = seg * 32 + i;
    out[(size_t)gm * DMODEL + col] = (xs[i] - mu) * rstd * gamma[col] + beta[col];
  }
}

// =====================================================================
extern "C" void kernel_launch(void* const* d_in, const int* in_sizes, int n_in,
                              void* d_out, int out_size, void* d_ws, size_t ws_size,
                              hipStream_t stream) {
  (void)in_sizes; (void)n_in; (void)out_size; (void)ws_size;
  const float* h     = (const float*)d_in[0];
  const float* Wq    = (const float*)d_in[1];
  const float* Wkv   = (const float*)d_in[2];
  const float* Wo    = (const float*)d_in[3];
  const float* gamma = (const float*)d_in[4];
  const float* beta  = (const float*)d_in[5];
  // d_in[6] = attn_mask (causal, baked into kernels)
  float* out = (float*)d_out;

  char* ws = (char*)d_ws;
  __bf16* Qb    = (__bf16*)(ws);                    // 4 MB  [b,h,s,d]
  __bf16* Kb    = (__bf16*)(ws + (4u  << 20));      // 4 MB  [b,h,s,d]
  __bf16* Vtb   = (__bf16*)(ws + (8u  << 20));      // 4 MB  [b,h,d,s]  (transposed)
  __bf16* attnb = (__bf16*)(ws + (12u << 20));      // 4 MB  [rows, 512]
  float*  stats = (float*)(ws + (16u << 20));       // 16*32*4160 f32 ~ 8.2 MB

  k_qkv<<<(ROWS / 16) * ((3 * DMODEL) / 64), 128, 0, stream>>>(h, Wq, Wkv, Qb, Kb, Vtb);
  k_chunk_state<<<BH * NCHUNK, 128, 0, stream>>>(Kb, Vtb, stats);
  k_attn<<<BH * NCHUNK, 128, 0, stream>>>(Qb, Kb, Vtb, stats, attnb);
  k_proj_ln<<<ROWS / 16, 256, 0, stream>>>(attnb, Wo, h, gamma, beta, out);
}
